// MultiHeadAttention_11871289606224
// MI455X (gfx1250) — compile-verified, matrix operand order checked by reading, not run on hardware
//
#include <hip/hip_runtime.h>

typedef __bf16 bf16;
typedef __attribute__((ext_vector_type(16))) __bf16 v16bf;
typedef __attribute__((ext_vector_type(8)))  __bf16 v8bf;
typedef __attribute__((ext_vector_type(4)))  __bf16 v4bf;
typedef __attribute__((ext_vector_type(8)))  float  v8f;
typedef __attribute__((ext_vector_type(4)))  float  v4f;

#define B_    2
#define T_    1024
#define CH_   1024
#define H_    16
#define KC_   64
#define WIN_  4
#define BAND_ 9

// ---------------------------------------------------------------------------
// WMMA fragment helpers (v_wmma_f32_16x16x32_bf16).
// 16-bit A-matrix 16x32 layout (ISA 7.12.2): lane m (0..15) holds row m,
// K = {0..7, 16..23}; lane m+16 holds row m, K = {8..15, 24..31}.
// B fragments load identically from the pre-transposed [N,K] operand.
// ---------------------------------------------------------------------------
union AFrag { v16bf v; v8bf h[2]; };

__device__ __forceinline__ AFrag load_frag(const bf16* __restrict__ base, int ld, int lane) {
  AFrag f;
  const bf16* p = base + (size_t)(lane & 15) * ld + ((lane & 16) ? 8 : 0);
  f.h[0] = *(const v8bf*)(p);
  f.h[1] = *(const v8bf*)(p + 16);
  return f;
}

// Shared 64x32 wave-tile K-loop: acc[4][2], 8 WMMAs per 6 fragment loads.
// A pre-offset to (M0, 0) with leading dim lda; Bt pre-offset to (N0, 0).
__device__ __forceinline__ void gemm_loop(const bf16* __restrict__ A, int lda,
                                          const bf16* __restrict__ Bt, int ldb,
                                          int K, int lane, v8f acc[4][2]) {
  for (int k0 = 0; k0 < K; k0 += 32) {
    AFrag a[4], b[2];
#pragma unroll
    for (int mi = 0; mi < 4; ++mi)
      a[mi] = load_frag(A + (size_t)(mi * 16) * lda + k0, lda, lane);
#pragma unroll
    for (int ni = 0; ni < 2; ++ni)
      b[ni] = load_frag(Bt + (size_t)(ni * 16) * ldb + k0, ldb, lane);
#pragma unroll
    for (int mi = 0; mi < 4; ++mi)
#pragma unroll
      for (int ni = 0; ni < 2; ++ni)
        acc[mi][ni] = __builtin_amdgcn_wmma_f32_16x16x32_bf16(
            false, a[mi].v, false, b[ni].v, (short)0, acc[mi][ni], false, false);
  }
}

// C/D 16x16 f32 layout: VGPR i -> row i (lanes 0-15) / row i+8 (lanes 16-31),
// col = lane & 15.

// ---------------------------------------------------------------------------
// Elementwise conversion, 4 elements per thread (b128 load, b64 store)
// ---------------------------------------------------------------------------
__global__ void k_f32_to_bf16(const float* __restrict__ in, bf16* __restrict__ out, int n4) {
  int i = blockIdx.x * blockDim.x + threadIdx.x;
  if (i < n4) {
    v4f v = ((const v4f*)in)[i];
    v4bf o;
#pragma unroll
    for (int j = 0; j < 4; ++j) o[j] = (bf16)v[j];
    ((v4bf*)out)[i] = o;
  }
}

// Wt[n][k] = W[k][n], 1024x1024, f32 -> bf16, LDS-tiled (coalesced both sides)
__global__ void k_transpose_bf16(const float* __restrict__ in, bf16* __restrict__ out) {
  __shared__ float tile[32][33];
  const int k0 = blockIdx.y * 32;
  const int n0 = blockIdx.x * 32;
  const int tx = threadIdx.x;   // 0..31
  const int ty = threadIdx.y;   // 0..7
#pragma unroll
  for (int j = 0; j < 4; ++j)
    tile[ty + j * 8][tx] = in[(size_t)(k0 + ty + j * 8) * CH_ + n0 + tx];
  __syncthreads();
#pragma unroll
  for (int j = 0; j < 4; ++j)
    out[(size_t)(n0 + ty + j * 8) * CH_ + k0 + tx] = (bf16)tile[tx][ty + j * 8];
}

// ---------------------------------------------------------------------------
// Projection GEMM: out(bf16) = (A[2048x1024] @ Bt^T + bias) * scale
// Block = 8 waves as 2(M)x4(N) of 64x32 wave tiles -> 128x128 block tile.
// mode 0: out[b][h][t][d]   (q with scale=1/8, k with scale=1)
// mode 2: out[b][h][d][t]   (v, pre-transposed for the PV GEMM B-operand)
// ---------------------------------------------------------------------------
__global__ void k_proj_gemm(const bf16* __restrict__ A, const bf16* __restrict__ Bt,
                            const float* __restrict__ bias, bf16* __restrict__ out,
                            int mode, float scale) {
  const int lane = threadIdx.x & 31;
  const int wave = threadIdx.x >> 5;
  const int wm = wave >> 2, wn = wave & 3;
  const int M0 = blockIdx.y * 128 + wm * 64;
  const int N0 = blockIdx.x * 128 + wn * 32;
  v8f acc[4][2] = {};
  gemm_loop(A + (size_t)M0 * CH_, CH_, Bt + (size_t)N0 * CH_, CH_, CH_, lane, acc);
  const int rl = (lane & 16) ? 8 : 0;
  const int cl = lane & 15;
  for (int mi = 0; mi < 4; ++mi)
    for (int ni = 0; ni < 2; ++ni) {
      int col = N0 + ni * 16 + cl;
      float bv = bias[col];
      int h = col >> 6, d = col & 63;
      for (int i = 0; i < 8; ++i) {
        int row = M0 + mi * 16 + rl + i;
        int b_ = row >> 10, t = row & 1023;
        float val = (acc[mi][ni][i] + bv) * scale;
        size_t oidx;
        if (mode == 2) oidx = ((size_t)(b_ * H_ + h) * KC_ + d) * T_ + t;
        else           oidx = ((size_t)(b_ * H_ + h) * T_ + t) * KC_ + d;
        out[oidx] = (bf16)val;
      }
    }
}

// ---------------------------------------------------------------------------
// scores[bh] = qs[bh] @ k[bh]^T   (f32 out, K = 64)
// ---------------------------------------------------------------------------
__global__ void k_scores_gemm(const bf16* __restrict__ qs, const bf16* __restrict__ kb,
                              float* __restrict__ scores) {
  const int bh = blockIdx.z;
  const int lane = threadIdx.x & 31;
  const int wave = threadIdx.x >> 5;
  const int wm = wave >> 2, wn = wave & 3;
  const int M0 = blockIdx.y * 128 + wm * 64;
  const int N0 = blockIdx.x * 128 + wn * 32;
  const bf16* Aq = qs + (size_t)bh * T_ * KC_;
  const bf16* Bk = kb + (size_t)bh * T_ * KC_;
  float* S = scores + (size_t)bh * T_ * T_;
  v8f acc[4][2] = {};
  gemm_loop(Aq + (size_t)M0 * KC_, KC_, Bk + (size_t)N0 * KC_, KC_, KC_, lane, acc);
  const int rl = (lane & 16) ? 8 : 0;
  const int cl = lane & 15;
  for (int mi = 0; mi < 4; ++mi)
    for (int ni = 0; ni < 2; ++ni)
      for (int i = 0; i < 8; ++i) {
        int row = M0 + mi * 16 + rl + i;
        int col = N0 + ni * 16 + cl;
        S[(size_t)row * T_ + col] = acc[mi][ni][i];
      }
}

// ---------------------------------------------------------------------------
// Banded relative-key logits: emb_rel_k has only 9 non-zero rows after
// _get_rel_emb zero-padding, so _rel_to_abs reduces to
//   scores[bh][t][t+dd-4] += qs[bh][t] . erk[dd]   for dd in [0,9)
// ---------------------------------------------------------------------------
__global__ void k_band_relk(const bf16* __restrict__ qs, const float* __restrict__ erk,
                            float* __restrict__ scores) {
  int idx = blockIdx.x * blockDim.x + threadIdx.x;
  if (idx >= B_ * H_ * T_ * BAND_) return;
  int dd = idx % BAND_;
  int t  = (idx / BAND_) % T_;
  int bh = idx / (BAND_ * T_);
  int j = t + dd - WIN_;
  if (j < 0 || j >= T_) return;
  const bf16* qrow = qs + ((size_t)bh * T_ + t) * KC_;
  const float* e = erk + dd * KC_;
  float s = 0.f;
  for (int kk = 0; kk < KC_; ++kk) s += (float)qrow[kk] * e[kk];
  scores[((size_t)bh * T_ + t) * T_ + j] += s;
}

// ---------------------------------------------------------------------------
// Masked softmax over each row; writes bf16 probabilities IN PLACE at the
// start of each f32 row (row pitch stays 4096 B => bf16 pitch 2048 elems).
// ---------------------------------------------------------------------------
__global__ void k_softmax(float* __restrict__ scores, const int* __restrict__ mask) {
  __shared__ float red[256];
  const int r  = blockIdx.x;            // 0 .. B*H*T-1
  const int bh = r >> 10;
  const int t  = r & (T_ - 1);
  const int b  = bh >> 4;
  float* row = scores + (size_t)r * T_;
  const int* mrow = mask + ((size_t)b * T_ + t) * T_;
  const int tid = threadIdx.x;
  float vals[4];
  float mx = -3.0e38f;
  for (int j = 0; j < 4; ++j) {
    int s = tid + j * 256;
    float v = row[s];
    if (mrow[s] == 0) v = 1e-4f;        // reference: masked BEFORE softmax
    vals[j] = v;
    mx = fmaxf(mx, v);
  }
  red[tid] = mx; __syncthreads();
  for (int off = 128; off > 0; off >>= 1) {
    if (tid < off) red[tid] = fmaxf(red[tid], red[tid + off]);
    __syncthreads();
  }
  mx = red[0]; __syncthreads();
  float sum = 0.f;
  for (int j = 0; j < 4; ++j) { vals[j] = __expf(vals[j] - mx); sum += vals[j]; }
  red[tid] = sum; __syncthreads();
  for (int off = 128; off > 0; off >>= 1) {
    if (tid < off) red[tid] += red[tid + off];
    __syncthreads();
  }
  float inv = 1.0f / red[0];
  bf16* prow = (bf16*)row;
  for (int j = 0; j < 4; ++j) {
    int s = tid + j * 256;
    prow[s] = (bf16)(vals[j] * inv);
  }
}

// ---------------------------------------------------------------------------
// out[bh] = p[bh] @ v[bh]  (+ fused banded rel-v term), bf16 out in [b,t,ch]
// p read as bf16 with pitch 2*T_ (in-place softmax output); v pre-transposed.
// Block = 8 waves as 4(M)x2(N) of 64x32 tiles -> 256x64 block tile (N = KC).
// ---------------------------------------------------------------------------
__global__ void k_pv_gemm(const float* __restrict__ scoresF, const bf16* __restrict__ vt,
                          const float* __restrict__ erv, bf16* __restrict__ attnb) {
  const int bh = blockIdx.z;
  const int lane = threadIdx.x & 31;
  const int wave = threadIdx.x >> 5;
  const int wm = wave >> 1, wn = wave & 1;
  const int M0 = blockIdx.y * 256 + wm * 64;
  const int N0 = wn * 32;                      // N = KC_ = 64
  const bf16* P = (const bf16*)(scoresF + (size_t)bh * T_ * T_);
  const bf16* V = vt + (size_t)bh * KC_ * T_;
  const int ldp = 2 * T_;
  v8f acc[4][2] = {};
  gemm_loop(P + (size_t)M0 * ldp, ldp, V + (size_t)N0 * T_, T_, T_, lane, acc);
  const int b_ = bh >> 4, h = bh & 15;
  const int rl = (lane & 16) ? 8 : 0;
  const int cl = lane & 15;
  for (int mi = 0; mi < 4; ++mi)
    for (int ni = 0; ni < 2; ++ni) {
      int d = N0 + ni * 16 + cl;
      for (int i = 0; i < 8; ++i) {
        int t = M0 + mi * 16 + rl + i;
        float val = acc[mi][ni][i];
        // fused _abs_to_rel + rel-v: out[t,d] += sum_dd p[t, t+dd-4] * erv[dd][d]
        const bf16* prow = P + (size_t)t * ldp;
        for (int dd = 0; dd < BAND_; ++dd) {
          int j = t + dd - WIN_;
          if (j >= 0 && j < T_) val += (float)prow[j] * erv[dd * KC_ + d];
        }
        attnb[((size_t)b_ * T_ + t) * CH_ + h * KC_ + d] = (bf16)val;
      }
    }
}

// ---------------------------------------------------------------------------
// Final projection: y(f32) = attnb @ Wo + bo
// ---------------------------------------------------------------------------
__global__ void k_final_gemm(const bf16* __restrict__ A, const bf16* __restrict__ Bt,
                             const float* __restrict__ bias, float* __restrict__ out) {
  const int lane = threadIdx.x & 31;
  const int wave = threadIdx.x >> 5;
  const int wm = wave >> 2, wn = wave & 3;
  const int M0 = blockIdx.y * 128 + wm * 64;
  const int N0 = blockIdx.x * 128 + wn * 32;
  v8f acc[4][2] = {};
  gemm_loop(A + (size_t)M0 * CH_, CH_, Bt + (size_t)N0 * CH_, CH_, CH_, lane, acc);
  const int rl = (lane & 16) ? 8 : 0;
  const int cl = lane & 15;
  for (int mi = 0; mi < 4; ++mi)
    for (int ni = 0; ni < 2; ++ni) {
      int col = N0 + ni * 16 + cl;
      float bv = bias[col];
      for (int i = 0; i < 8; ++i) {
        int row = M0 + mi * 16 + rl + i;
        out[(size_t)row * CH_ + col] = acc[mi][ni][i] + bv;
      }
    }
}

// ---------------------------------------------------------------------------
extern "C" void kernel_launch(void* const* d_in, const int* in_sizes, int n_in,
                              void* d_out, int out_size, void* d_ws, size_t ws_size,
                              hipStream_t stream) {
  const float* x    = (const float*)d_in[0];
  const float* c    = (const float*)d_in[1];
  const int*   mask = (const int*)  d_in[2];
  const float* Wq   = (const float*)d_in[3];
  const float* bq   = (const float*)d_in[4];
  const float* Wk   = (const float*)d_in[5];
  const float* bk   = (const float*)d_in[6];
  const float* Wv   = (const float*)d_in[7];
  const float* bv   = (const float*)d_in[8];
  const float* Wo   = (const float*)d_in[9];
  const float* bo   = (const float*)d_in[10];
  const float* erk  = (const float*)d_in[11];
  const float* erv  = (const float*)d_in[12];
  float* y = (float*)d_out;

  char* ws = (char*)d_ws;
  const size_t MB = 1024ull * 1024ull;
  bf16*  xb     = (bf16*) (ws +  0 * MB);   // [2048,1024]
  bf16*  cb     = (bf16*) (ws +  4 * MB);   // [2048,1024]
  bf16*  Wqt    = (bf16*) (ws +  8 * MB);   // [N,K] transposed
  bf16*  Wkt    = (bf16*) (ws + 10 * MB);
  bf16*  Wvt    = (bf16*) (ws + 12 * MB);
  bf16*  Wot    = (bf16*) (ws + 14 * MB);
  bf16*  qs     = (bf16*) (ws + 16 * MB);   // [B,H,T,KC], pre-scaled by 1/8
  bf16*  kb_    = (bf16*) (ws + 20 * MB);   // [B,H,T,KC]
  bf16*  vt     = (bf16*) (ws + 24 * MB);   // [B,H,KC,T]
  bf16*  attnb  = (bf16*) (ws + 28 * MB);   // [B,T,CH]
  float* scores = (float*)(ws + 32 * MB);   // [B*H,T,T] f32, reused as bf16 p

  const int NE4 = (B_ * T_ * CH_) / 4;      // 524,288 quads
  k_f32_to_bf16<<<NE4 / 256, 256, 0, stream>>>(x, xb, NE4);
  k_f32_to_bf16<<<NE4 / 256, 256, 0, stream>>>(c, cb, NE4);

  dim3 tg(32, 32), tb(32, 8);
  k_transpose_bf16<<<tg, tb, 0, stream>>>(Wq, Wqt);
  k_transpose_bf16<<<tg, tb, 0, stream>>>(Wk, Wkt);
  k_transpose_bf16<<<tg, tb, 0, stream>>>(Wv, Wvt);
  k_transpose_bf16<<<tg, tb, 0, stream>>>(Wo, Wot);

  dim3 gp(CH_ / 128, (B_ * T_) / 128, 1);   // 8 x 16
  k_proj_gemm<<<gp, 256, 0, stream>>>(xb, Wqt, bq, qs,  0, 0.125f); // q, scaled
  k_proj_gemm<<<gp, 256, 0, stream>>>(cb, Wkt, bk, kb_, 0, 1.0f);   // k
  k_proj_gemm<<<gp, 256, 0, stream>>>(cb, Wvt, bv, vt,  2, 1.0f);   // v (transposed)

  dim3 gs(T_ / 128, T_ / 128, B_ * H_);     // 8 x 8 x 32
  k_scores_gemm<<<gs, 256, 0, stream>>>(qs, kb_, scores);

  int nband = B_ * H_ * T_ * BAND_;
  k_band_relk<<<(nband + 255) / 256, 256, 0, stream>>>(qs, erk, scores);

  k_softmax<<<B_ * H_ * T_, 256, 0, stream>>>(scores, mask);

  dim3 gv(1, T_ / 256, B_ * H_);            // 1 x 4 x 32
  k_pv_gemm<<<gv, 256, 0, stream>>>(scores, vt, erv, attnb);

  k_final_gemm<<<gp, 256, 0, stream>>>(attnb, Wot, bo, y);
}
